// PropertyPredictor_17995912970762
// MI455X (gfx1250) — compile-verified, matrix operand order checked by reading, not run on hardware
//
#include <hip/hip_runtime.h>
#include <hip/hip_bf16.h>
#include <math.h>

typedef __attribute__((ext_vector_type(16))) _Float16 v16h;
typedef __attribute__((ext_vector_type(8)))  float    v8f;

#define B_   16
#define N_   256
#define F_   64
#define FP_  16
#define H_   128

#define LOG2E_ 1.4426950408889634f
#define LN2_   0.6931471805599453f

// Stable softplus for unbounded args (stage-2 only; few elements).
__device__ __forceinline__ float softplus_stable(float x) {
    float t = __builtin_amdgcn_exp2f(-fabsf(x) * LOG2E_);
    float l = __builtin_amdgcn_logf(1.0f + t);
    return fmaf(LN2_, l, fmaxf(x, 0.0f));
}

// log2-domain softplus for provably bounded args:
// returns log2(1 + 2^y) = log2e * softplus(y/log2e).  Just exp2+add+log2.
__device__ __forceinline__ float softplus_l2(float y) {
    return __builtin_amdgcn_logf(1.0f + __builtin_amdgcn_exp2f(y));
}

__global__ __launch_bounds__(256)
void pp_fused_kernel(const float* __restrict__ r,    // [B,N,3]
                     const float* __restrict__ fb,   // [B,N,F]
                     const float* __restrict__ w1,   // [1,H]
                     const float* __restrict__ b1,   // [H]
                     const float* __restrict__ w2,   // [H,F]
                     const float* __restrict__ b2,   // [F]
                     const float* __restrict__ pw1,  // [F,H]
                     const float* __restrict__ pb1,  // [H]
                     const float* __restrict__ pw2,  // [H,FP]
                     const float* __restrict__ pb2,  // [FP]
                     float* __restrict__ out)        // [B,N,FP]
{
    __shared__ float s_d[N_];
    __shared__ float s_w1[H_];    // pre-scaled by log2e
    __shared__ float s_b1[H_];    // pre-scaled by log2e
    __shared__ float s_b2[F_];    // pre-scaled by log2e
    __shared__ alignas(32) _Float16 s_bfrag[16 * 32 * 16]; // [kstep*4+ntile][lane][16]
    __shared__ float s_fbar[F_];
    __shared__ float s_fb2[F_];
    __shared__ float s_h2[H_];

    const int tid = threadIdx.x;
    const int wg  = blockIdx.x;        // 0 .. B*N-1, one (b,i) per workgroup
    const int b   = wg >> 8;
    const int i   = wg & 255;

    // ---------------- prep phase ----------------
    {
        const float xi = r[(b * N_ + i) * 3 + 0];
        const float yi = r[(b * N_ + i) * 3 + 1];
        const float zi = r[(b * N_ + i) * 3 + 2];
        const int j = tid;
        const float dx = xi - r[(b * N_ + j) * 3 + 0];
        const float dy = yi - r[(b * N_ + j) * 3 + 1];
        const float dz = zi - r[(b * N_ + j) * 3 + 2];
        const float sq = dx * dx + dy * dy + dz * dz;
        s_d[j] = (sq > 0.0f) ? sqrtf(sq) : 0.0f;
    }
    // Fold log2e into the hidden-layer affine and into b2 (see derivation):
    //  A stores l = log2(1+2^(log2e*x)) = log2e * hidden
    //  acc (A x w2)  = log2e * (hidden @ w2);  y = acc + log2e*b2
    //  l2 = log2(1+2^y) = log2e * d_prime;  single ln2 applied after j-reduce.
    if (tid < H_) { s_w1[tid] = w1[tid] * LOG2E_; s_b1[tid] = b1[tid] * LOG2E_; }
    if (tid < F_) { s_b2[tid] = b2[tid] * LOG2E_; s_fbar[tid] = 0.0f; }

    // Pre-swizzle mp_w2 (H x F, f32) into per-lane f16 B-fragments (raw w2),
    // K ordering mirrors the documented 16-bit A layout.
    for (int e = tid; e < 512; e += 256) {            // 16 frags * 32 lanes
        const int fi    = e >> 5;
        const int lane  = e & 31;
        const int kstep = fi >> 2;
        const int ntile = fi & 3;
        const int n     = ntile * 16 + (lane & 15);
        const int kb    = kstep * 32 + (lane >> 4) * 8;
        _Float16* dst = &s_bfrag[(fi * 32 + lane) * 16];
#pragma unroll
        for (int p = 0; p < 2; ++p)
#pragma unroll
            for (int q = 0; q < 8; ++q) {
                const int k = kb + p * 16 + q;
                dst[p * 8 + q] = (_Float16)w2[k * F_ + n];
            }
    }
    __syncthreads();

    // ---------------- WMMA GEMM: [256 x 128] x [128 x 64] ----------------
    const int wave  = tid >> 5;        // 0..7, owns rows [32*wave, 32*wave+32)
    const int lane  = tid & 31;
    const int half8 = (lane >> 4) * 8; // K sub-offset per lane group
    const int lrow  = lane & 15;       // M (A) / N (B,C) index within tile

    v8f zero = {};
    v8f acc[2][4];
#pragma unroll
    for (int m = 0; m < 2; ++m)
#pragma unroll
        for (int n = 0; n < 4; ++n)
            acc[m][n] = zero;

    float dr[2];
    dr[0] = s_d[wave * 32 + lrow];
    dr[1] = s_d[wave * 32 + 16 + lrow];

#pragma unroll
    for (int s = 0; s < 4; ++s) {
        // B fragments for this K-step, shared across both M-tiles
        v16h bfr[4];
#pragma unroll
        for (int n = 0; n < 4; ++n)
            bfr[n] = *reinterpret_cast<const v16h*>(&s_bfrag[((s * 4 + n) * 32 + lane) * 16]);

        const int kb = s * 32 + half8;
#pragma unroll
        for (int m = 0; m < 2; ++m) {
            // A fragment: l = log2(1 + 2^(log2e*(d*w1+b1))); arg bounded (|x|<=2.74)
            v16h a;
#pragma unroll
            for (int p = 0; p < 2; ++p)
#pragma unroll
                for (int q = 0; q < 8; ++q) {
                    const int h = kb + p * 16 + q;
                    const float y = fmaf(dr[m], s_w1[h], s_b1[h]);  // log2-domain
                    a[p * 8 + q] = (_Float16)softplus_l2(y);
                }
#pragma unroll
            for (int n = 0; n < 4; ++n)
                acc[m][n] = __builtin_amdgcn_wmma_f32_16x16x32_f16(
                    false, a, false, bfr[n], (short)0, acc[m][n], false, false);
        }
    }

    // ---------------- epilogue: l2 = log2-softplus(acc + log2e*b2); reduce over j ----
    // C layout: VGPR v, lane l -> M = v + 8*(l>=16), N = l%16
#pragma unroll
    for (int n = 0; n < 4; ++n) {
        const int fcol = n * 16 + lrow;
        const float bias = s_b2[fcol];
        float partial = 0.0f;
#pragma unroll
        for (int m = 0; m < 2; ++m) {
#pragma unroll
            for (int v = 0; v < 8; ++v) {
                const int row = wave * 32 + m * 16 + v + half8;
                const float l2 = softplus_l2(acc[m][n][v] + bias); // |arg| <= ~46
                partial = fmaf(l2, fb[(b * N_ + row) * F_ + fcol], partial);
            }
        }
        atomicAdd(&s_fbar[fcol], partial);   // accumulates log2e * sum(d_prime*f)
    }
    __syncthreads();

    // ---------------- fused pointwise MLP: F -> 128 -> FP ----------------
    if (tid < F_) s_fb2[tid] = softplus_stable(LN2_ * s_fbar[tid]); // undo log2e once
    __syncthreads();
    if (tid < H_) {
        float a2 = pb1[tid];
#pragma unroll 8
        for (int ff = 0; ff < F_; ++ff)
            a2 = fmaf(s_fb2[ff], pw1[ff * H_ + tid], a2);
        s_h2[tid] = softplus_stable(a2);
    }
    __syncthreads();
    if (tid < FP_) {
        float a3 = pb2[tid];
#pragma unroll 8
        for (int h = 0; h < H_; ++h)
            a3 = fmaf(s_h2[h], pw2[h * FP_ + tid], a3);
        out[wg * FP_ + tid] = softplus_stable(a3);
    }
}

extern "C" void kernel_launch(void* const* d_in, const int* in_sizes, int n_in,
                              void* d_out, int out_size, void* d_ws, size_t ws_size,
                              hipStream_t stream) {
    (void)in_sizes; (void)n_in; (void)out_size; (void)d_ws; (void)ws_size;
    const float* r   = (const float*)d_in[0];
    const float* fbp = (const float*)d_in[1];
    const float* w1  = (const float*)d_in[2];
    const float* b1  = (const float*)d_in[3];
    const float* w2  = (const float*)d_in[4];
    const float* b2  = (const float*)d_in[5];
    const float* pw1 = (const float*)d_in[6];
    const float* pb1 = (const float*)d_in[7];
    const float* pw2 = (const float*)d_in[8];
    const float* pb2 = (const float*)d_in[9];
    float* out = (float*)d_out;

    dim3 grid(B_ * N_);   // 4096 workgroups, one per (b, i)
    dim3 block(256);      // 8 wave32
    pp_fused_kernel<<<grid, block, 0, stream>>>(r, fbp, w1, b1, w2, b2,
                                                pw1, pb1, pw2, pb2, out);
}
